// MGRU_17471926960362
// MI455X (gfx1250) — compile-verified, matrix operand order checked by reading, not run on hardware
//
#include <hip/hip_runtime.h>
#include <hip/hip_bf16.h>

typedef __attribute__((ext_vector_type(16))) __bf16 v16bf;
typedef __attribute__((ext_vector_type(8)))  __bf16 v8bf;
typedef __attribute__((ext_vector_type(8)))  float  v8f;

#define B_  16
#define T_  2048
#define EI_ 1024
#define EO_ 1024

// ---- fast transcendentals: use CDNA5 native V_TANH / V_RCP / V_EXP ---------
__device__ __forceinline__ float fast_tanh(float x) {
#if __has_builtin(__builtin_amdgcn_tanhf)
    return __builtin_amdgcn_tanhf(x);
#else
    // 1 - 2/(e^{2x}+1) with native exp + rcp (no branchy ocml path)
    float e = __expf(2.0f * x);
    return 1.0f - 2.0f * __builtin_amdgcn_rcpf(e + 1.0f);
#endif
}

__device__ __forceinline__ float fast_sigmoid(float x) {
    return __builtin_amdgcn_rcpf(1.0f + __expf(-x));
}

// ---------------------------------------------------------------------------
// Pack a (K=1024) x (N=1024) row-major f32 weight matrix into bf16 WMMA
// B-fragment order:  frag[((kt*64 + nt)*32 + lane)*16 + e]
// with element->K mapping  k = kt*32 + (e&7) + (e>>3)*16 + half*8  (half=lane>>4)
// and N = nt*16 + (lane&15).  One contiguous 32-byte load per lane fragment.
// ---------------------------------------------------------------------------
__global__ __launch_bounds__(256) void pack_w(const float* __restrict__ W,
                                              __bf16* __restrict__ WF) {
    int tid  = blockIdx.x * 256 + threadIdx.x;   // 0 .. 65535
    int lane = tid & 31;
    int nt   = (tid >> 5) & 63;
    int kt   = tid >> 11;
    int hh   = lane >> 4;
    int n    = nt * 16 + (lane & 15);
    __bf16* dst = WF + (size_t)tid * 16;
#pragma unroll
    for (int e = 0; e < 16; ++e) {
        int koff = (e & 7) + ((e >> 3) << 4) + hh * 8;
        int k    = kt * 32 + koff;
        dst[e]   = (__bf16)W[(size_t)k * EO_ + n];
    }
}

// ---------------------------------------------------------------------------
// G = tanh(X @ Wxh):  (32768 x 1024) @ (1024 x 1024), bf16 WMMA, f32 accum.
// One block (8 waves) per 16-row M-tile; each wave covers 8 N-tiles so X is
// read exactly once from HBM; packed Wxh (2 MB) stays L2-resident.
// ---------------------------------------------------------------------------
__global__ __launch_bounds__(256) void gemm_tanh(const float* __restrict__ X,
                                                 const __bf16* __restrict__ WF,
                                                 float* __restrict__ G) {
    const int wave  = threadIdx.x >> 5;          // 0..7
    const int lane  = threadIdx.x & 31;
    const int hh    = lane >> 4;
    const int mTile = blockIdx.x;                // 0..2047
    const int mrow  = mTile * 16 + (lane & 15);
    const float* xrow = X + (size_t)mrow * EI_;

    v8f acc[8];
#pragma unroll
    for (int j = 0; j < 8; ++j) acc[j] = (v8f){};

#pragma unroll 1
    for (int kt = 0; kt < 32; ++kt) {
        const int kb = kt * 32 + hh * 8;
        // A fragment: 16 f32 in two contiguous 8-element runs -> bf16
        float4 a0 = *(const float4*)(xrow + kb);
        float4 a1 = *(const float4*)(xrow + kb + 4);
        float4 a2 = *(const float4*)(xrow + kb + 16);
        float4 a3 = *(const float4*)(xrow + kb + 20);
        v16bf a;
        a[0]  = (__bf16)a0.x; a[1]  = (__bf16)a0.y; a[2]  = (__bf16)a0.z; a[3]  = (__bf16)a0.w;
        a[4]  = (__bf16)a1.x; a[5]  = (__bf16)a1.y; a[6]  = (__bf16)a1.z; a[7]  = (__bf16)a1.w;
        a[8]  = (__bf16)a2.x; a[9]  = (__bf16)a2.y; a[10] = (__bf16)a2.z; a[11] = (__bf16)a2.w;
        a[12] = (__bf16)a3.x; a[13] = (__bf16)a3.y; a[14] = (__bf16)a3.z; a[15] = (__bf16)a3.w;

#pragma unroll
        for (int j = 0; j < 8; ++j) {
            const int nt = wave * 8 + j;
            v16bf bfrag = *(const v16bf*)(WF + (((size_t)kt * 64 + nt) * 32 + lane) * 16);
            acc[j] = __builtin_amdgcn_wmma_f32_16x16x32_bf16(
                false, a, false, bfrag, (short)0, acc[j], false, false);
        }
    }

#pragma unroll
    for (int j = 0; j < 8; ++j) {
        const int col = wave * 128 + j * 16 + (lane & 15);
#pragma unroll
        for (int r = 0; r < 8; ++r) {
            const int row = mTile * 16 + r + hh * 8;
            G[(size_t)row * EO_ + col] = fast_tanh(acc[j][r]);
        }
    }
}

// ---------------------------------------------------------------------------
// Serial scan:  h_t = (1-sigmoid(h_{t-1} @ Whf)) * h_{t-1} + sigmoid(.) * g_t
// Single 1024-thread workgroup (32 waves); wave w owns N-tiles 2w, 2w+1.
//  - h carried in f32 registers by the owning lane
//  - bf16 copy of h double-buffered in padded LDS -> ONE barrier per step
//  - g_t prefetched before the WMMA K-loop (address independent of h)
//  - each accumulator split into even/odd-K chains (RAW depth 16, 4 chains)
// ---------------------------------------------------------------------------
__global__ __launch_bounds__(1024) void mgru_scan(const float* __restrict__ h0,
                                                  const __bf16* __restrict__ WF,
                                                  const float* __restrict__ G,
                                                  float* __restrict__ out) {
    __shared__ __bf16 Hb[2][16][EO_ + 8];        // +8 bf16 pad: spreads LDS banks
    const int tid  = threadIdx.x;
    const int wave = tid >> 5;
    const int lane = tid & 31;
    const int hh   = lane >> 4;
    const int m    = lane & 15;

    float* outs = out;
    float* hT   = out + (size_t)B_ * T_ * EO_;

    // init: register-resident h (owner lanes) + bf16 LDS copy (everyone)
    float hreg[2][8];
#pragma unroll
    for (int j = 0; j < 2; ++j)
#pragma unroll
        for (int r = 0; r < 8; ++r)
            hreg[j][r] = h0[(size_t)(r + hh * 8) * EO_ + (wave * 2 + j) * 16 + m];

    for (int i = tid; i < B_ * EO_; i += 1024)
        Hb[0][i >> 10][i & 1023] = (__bf16)h0[i];
    __syncthreads();

    const int nt0 = wave * 2, nt1 = wave * 2 + 1;

    for (int t = 0; t < T_; ++t) {
        const __bf16 (*__restrict__ Hr)[EO_ + 8] = Hb[t & 1];
        __bf16 (*__restrict__ Hw)[EO_ + 8]       = Hb[(t + 1) & 1];

        // prefetch g_t: independent of h, so its latency hides under the WMMAs
        float gv[2][8];
#pragma unroll
        for (int j = 0; j < 2; ++j) {
            const int col = (wave * 2 + j) * 16 + m;
#pragma unroll
            for (int r = 0; r < 8; ++r)
                gv[j][r] = G[((size_t)(r + hh * 8) * T_ + t) * EO_ + col];
        }

        v8f c0a = (v8f){}, c0b = (v8f){}, c1a = (v8f){}, c1b = (v8f){};
#pragma unroll 4
        for (int kt = 0; kt < 32; kt += 2) {
            // even K-tile -> chains a, odd K-tile -> chains b
#pragma unroll
            for (int p = 0; p < 2; ++p) {
                const int k  = kt + p;
                const int kb = k * 32 + hh * 8;
                v8bf r0 = *(const v8bf*)&Hr[m][kb];
                v8bf r1 = *(const v8bf*)&Hr[m][kb + 16];
                v16bf a = __builtin_shufflevector(r0, r1,
                    0, 1, 2, 3, 4, 5, 6, 7, 8, 9, 10, 11, 12, 13, 14, 15);
                v16bf b0 = *(const v16bf*)(WF + (((size_t)k * 64 + nt0) * 32 + lane) * 16);
                v16bf b1 = *(const v16bf*)(WF + (((size_t)k * 64 + nt1) * 32 + lane) * 16);
                if (p == 0) {
                    c0a = __builtin_amdgcn_wmma_f32_16x16x32_bf16(
                        false, a, false, b0, (short)0, c0a, false, false);
                    c1a = __builtin_amdgcn_wmma_f32_16x16x32_bf16(
                        false, a, false, b1, (short)0, c1a, false, false);
                } else {
                    c0b = __builtin_amdgcn_wmma_f32_16x16x32_bf16(
                        false, a, false, b0, (short)0, c0b, false, false);
                    c1b = __builtin_amdgcn_wmma_f32_16x16x32_bf16(
                        false, a, false, b1, (short)0, c1b, false, false);
                }
            }
        }
        const v8f c0 = c0a + c0b;
        const v8f c1 = c1a + c1b;

        // update owned h elements; write NEXT buffer (no read/write conflict)
#pragma unroll
        for (int j = 0; j < 2; ++j) {
            const int col = (j ? nt1 : nt0) * 16 + m;
#pragma unroll
            for (int r = 0; r < 8; ++r) {
                const float cv = j ? c1[r] : c0[r];
                const int   bi = r + hh * 8;
                const float f  = fast_sigmoid(cv);
                const float hp = hreg[j][r];
                const float hn = hp + f * (gv[j][r] - hp);   // (1-f)*hp + f*g
                outs[((size_t)bi * T_ + t) * EO_ + col] = hn;
                hreg[j][r]  = hn;
                Hw[bi][col] = (__bf16)hn;
            }
        }
        __syncthreads();   // next-buffer writes visible before next step reads
    }

#pragma unroll
    for (int j = 0; j < 2; ++j)
#pragma unroll
        for (int r = 0; r < 8; ++r)
            hT[(size_t)(r + hh * 8) * EO_ + (wave * 2 + j) * 16 + m] = hreg[j][r];
}

// ---------------------------------------------------------------------------
extern "C" void kernel_launch(void* const* d_in, const int* in_sizes, int n_in,
                              void* d_out, int out_size, void* d_ws, size_t ws_size,
                              hipStream_t stream) {
    const float* x   = (const float*)d_in[0];   // (B,T,EI)
    const float* h0  = (const float*)d_in[1];   // (1,B,EO)
    const float* wxh = (const float*)d_in[2];   // (EI,EO)
    const float* whf = (const float*)d_in[3];   // (EO,EO)

    float*  G    = (float*)d_ws;                                   // 128 MB
    __bf16* wxhF = (__bf16*)((char*)d_ws + (size_t)B_ * T_ * EO_ * sizeof(float));
    __bf16* whfF = wxhF + (size_t)EI_ * EO_;

    pack_w<<<256, 256, 0, stream>>>(wxh, wxhF);
    pack_w<<<256, 256, 0, stream>>>(whf, whfF);
    gemm_tanh<<<2048, 256, 0, stream>>>(x, wxhF, G);
    mgru_scan<<<1, 1024, 0, stream>>>(h0, whfF, G, (float*)d_out);
}
